// iResidualLayer_35158602285651
// MI455X (gfx1250) — matrix operand order repaired and checked by reading, not compiled, not run on hardware
//
#include <hip/hip_runtime.h>
#include <hip/hip_bf16.h>
#include <stdint.h>
#include <stddef.h>

// ---------------------------------------------------------------------------
// Problem constants (from reference): B=8192, D=1024, H=4096
// ---------------------------------------------------------------------------
#define BQ 8192
#define DQ 1024
#define HQ 4096

typedef __attribute__((ext_vector_type(16))) __bf16        bf16x16;
typedef __attribute__((ext_vector_type(8)))  float         f32x8;
typedef __attribute__((ext_vector_type(4)))  unsigned int  u32x4;
typedef int v4i_vs __attribute__((vector_size(16)));   // matches builtin param type

// GEMM tiling: 128(M) x 256(N) block tile, K-step 32, 256 threads = 8 wave32
// waves (2 in M x 4 in N); each wave computes a 64x64 tile = 4x4 WMMA 16x16
// accumulators -> 16 v_wmma per K-tile per wave.
#define BM 128
#define BN 256
#define BKK 32
#define GEMM_THREADS 256

// CDNA5 async global->LDS path (ASYNCcnt-tracked DMA), with safe fallback.
#if defined(__gfx1250__) && __has_builtin(__builtin_amdgcn_global_load_async_to_lds_b128)
#define USE_ASYNC_LDS 1
#else
#define USE_ASYNC_LDS 0
#endif

#if USE_ASYNC_LDS
__device__ __forceinline__ void async_copy16(const void* g, void* l) {
    __builtin_amdgcn_global_load_async_to_lds_b128(
        (__attribute__((address_space(1))) v4i_vs*)(void*)g,
        (__attribute__((address_space(3))) v4i_vs*)l,
        0, 0);
}
__device__ __forceinline__ void wait_async0() {
#if __has_builtin(__builtin_amdgcn_s_wait_asynccnt)
    __builtin_amdgcn_s_wait_asynccnt(0);
#else
    asm volatile("s_wait_asynccnt 0x0" ::: "memory");
#endif
}
#endif

// Epilogue modes
//  0: out_bf = relu(acc + bias[col]); maskOut = (acc+bias)>0          (fwd L1/L2)
//  1: outF   = acc + bias[col] + residual[idx]                        (fwd L3 -> y)
//  2: out_bf = acc * maskIn[idx]                                      (vjp L1/L2)
//  3: outF   = acc; out_bf = acc                                      (vjp L3 -> ut)
__global__ __launch_bounds__(GEMM_THREADS)
void gemm_bf16_wmma(const __bf16* __restrict__ A,    // [M,K] row-major
                    const __bf16* __restrict__ Bt,   // [N,K] row-major (C = A * Bt^T)
                    int M, int N, int K,
                    const float* __restrict__ bias,
                    const float* __restrict__ residual,
                    const unsigned char* __restrict__ maskIn,
                    unsigned char* __restrict__ maskOut,
                    __bf16* __restrict__ outBf,
                    float*  __restrict__ outF,
                    int mode)
{
    __shared__ __align__(16) __bf16 sA[2][BM * BKK];   // 2 x 8 KB
    __shared__ __align__(16) __bf16 sB[2][BN * BKK];   // 2 x 16 KB

    const int tid   = threadIdx.x;
    const int lane  = tid & 31;
    const int wave  = tid >> 5;       // 0..7
    const int waveM = wave & 1;       // 2 waves tile M (64 rows each)
    const int waveN = wave >> 1;      // 4 waves tile N (64 cols each)
    const int kh    = lane >> 4;      // lane half (0/1)
    const int lr    = lane & 15;

    const int blockN0 = blockIdx.x * BN;
    const int blockM0 = blockIdx.y * BM;

    const f32x8 zero8 = {};
    f32x8 acc[4][4];
#pragma unroll
    for (int i = 0; i < 4; ++i)
#pragma unroll
        for (int j = 0; j < 4; ++j) acc[i][j] = zero8;

    const int kTiles = K / BKK;

    // ---- compute on one LDS buffer -------------------------------------
    auto compute = [&](int buf) {
        const __bf16* pA = sA[buf];
        const __bf16* pB = sB[buf];
        // B fragment 32x16 bf16: lane half h holds K = h*16..h*16+15 contiguous.
        bf16x16 bF[4];
#pragma unroll
        for (int ni = 0; ni < 4; ++ni) {
            const __bf16* base = &pB[(waveN * 64 + ni * 16 + lr) * BKK];
            union { u32x4 q[2]; bf16x16 v; } fr;
            fr.q[0] = *(const u32x4*)(base + kh * 16);
            fr.q[1] = *(const u32x4*)(base + kh * 16 + 8);
            bF[ni] = fr.v;
        }
        // A fragment 16x32 bf16: lane half h holds K = h*16..h*16+15 contiguous.
#pragma unroll
        for (int mi = 0; mi < 4; ++mi) {
            const __bf16* base = &pA[(waveM * 64 + mi * 16 + lr) * BKK];
            union { u32x4 q[2]; bf16x16 v; } fr;
            fr.q[0] = *(const u32x4*)(base + kh * 16);
            fr.q[1] = *(const u32x4*)(base + kh * 16 + 8);
            bf16x16 aF = fr.v;
#pragma unroll
            for (int ni = 0; ni < 4; ++ni)
                acc[mi][ni] = __builtin_amdgcn_wmma_f32_16x16x32_bf16(
                    false, aF, false, bF[ni], (short)0, acc[mi][ni],
                    false, false);
        }
    };

#if USE_ASYNC_LDS
    // ---- async DMA double-buffered pipeline ----------------------------
    auto stage = [&](int kt, int b) {
        const int k0 = kt * BKK;
#pragma unroll
        for (int s = 0; s < 2; ++s) {            // A: 128x32 bf16 = 512 x 16B
            int c = tid + s * GEMM_THREADS;
            int row = c >> 2, col8 = (c & 3) * 8;
            async_copy16(&A[(size_t)(blockM0 + row) * K + k0 + col8],
                         &sA[b][row * BKK + col8]);
        }
#pragma unroll
        for (int s = 0; s < 4; ++s) {            // B: 256x32 bf16 = 1024 x 16B
            int c = tid + s * GEMM_THREADS;
            int row = c >> 2, col8 = (c & 3) * 8;
            async_copy16(&Bt[(size_t)(blockN0 + row) * K + k0 + col8],
                         &sB[b][row * BKK + col8]);
        }
    };

    stage(0, 0);
    for (int kt = 0; kt < kTiles; ++kt) {
        const int buf = kt & 1;
        // Wait for this wave's async loads of tile kt (loads complete in order).
        wait_async0();
        // All waves: tile kt resident AND everyone done reading buf^1.
        __syncthreads();
        // Overlap: DMA tile kt+1 into the other buffer while we do WMMAs.
        if (kt + 1 < kTiles) stage(kt + 1, buf ^ 1);
        compute(buf);
    }
#else
    // ---- fallback: VGPR-staged single buffer ---------------------------
    for (int kt = 0; kt < kTiles; ++kt) {
        const int k0 = kt * BKK;
#pragma unroll
        for (int s = 0; s < 2; ++s) {
            int c = tid + s * GEMM_THREADS;
            int row = c >> 2, col8 = (c & 3) * 8;
            *(u32x4*)&sA[0][row * BKK + col8] =
                *(const u32x4*)&A[(size_t)(blockM0 + row) * K + k0 + col8];
        }
#pragma unroll
        for (int s = 0; s < 4; ++s) {
            int c = tid + s * GEMM_THREADS;
            int row = c >> 2, col8 = (c & 3) * 8;
            *(u32x4*)&sB[0][row * BKK + col8] =
                *(const u32x4*)&Bt[(size_t)(blockN0 + row) * K + k0 + col8];
        }
        __syncthreads();
        if (kt + 1 < kTiles) {
            int row = tid >> 2, col8 = (tid & 3) * 8;
            __builtin_prefetch(&A[(size_t)(blockM0 + row) * K + k0 + BKK + col8], 0, 3);
            __builtin_prefetch(&Bt[(size_t)(blockN0 + row) * K + k0 + BKK + col8], 0, 3);
        }
        compute(0);
        __syncthreads();
    }
#endif

    // Epilogue. C layout: lanes 0-15 -> N=lane, M=vgpr; lanes 16-31 -> N=lane-16, M=vgpr+8
#pragma unroll
    for (int mi = 0; mi < 4; ++mi) {
#pragma unroll
        for (int ni = 0; ni < 4; ++ni) {
#pragma unroll
            for (int r = 0; r < 8; ++r) {
                int row = blockM0 + waveM * 64 + mi * 16 + kh * 8 + r;
                int col = blockN0 + waveN * 64 + ni * 16 + lr;
                size_t idx = (size_t)row * N + col;
                float v = acc[mi][ni][r];
                if (mode == 0) {
                    v += bias[col];
                    unsigned char m = v > 0.0f ? 1 : 0;
                    maskOut[idx] = m;
                    outBf[idx] = (__bf16)(m ? v : 0.0f);
                } else if (mode == 1) {
                    outF[idx] = v + bias[col] + residual[idx];
                } else if (mode == 2) {
                    v *= (float)maskIn[idx];
                    outBf[idx] = (__bf16)v;
                } else {
                    outF[idx] = v;
                    outBf[idx] = (__bf16)v;
                }
            }
        }
    }
}

// fp32 -> bf16 elementwise
__global__ void convert_f32_bf16(const float* __restrict__ in,
                                 __bf16* __restrict__ out, size_t n) {
    size_t i = (size_t)blockIdx.x * blockDim.x + threadIdx.x;
    size_t stride = (size_t)gridDim.x * blockDim.x;
    for (; i < n; i += stride) out[i] = (__bf16)in[i];
}

// out[C][R] = (bf16) in[R][C]   (R, C multiples of 32)
__global__ __launch_bounds__(256)
void transpose_f32_to_bf16(const float* __restrict__ in,
                           __bf16* __restrict__ out, int R, int C) {
    __shared__ float tile[32][33];
    int c0 = blockIdx.x * 32, r0 = blockIdx.y * 32;
    int tx = threadIdx.x, ty = threadIdx.y;  // 32 x 8
#pragma unroll
    for (int i = 0; i < 32; i += 8)
        tile[ty + i][tx] = in[(size_t)(r0 + ty + i) * C + c0 + tx];
    __syncthreads();
#pragma unroll
    for (int i = 0; i < 32; i += 8)
        out[(size_t)(c0 + ty + i) * R + r0 + tx] = (__bf16)tile[tx][ty + i];
}

__global__ void zero_f32(float* __restrict__ p, int n) {
    int i = blockIdx.x * 256 + threadIdx.x;
    if (i < n) p[i] = 0.0f;
}

// out[row] += coeff * dot(a[row,:], b[row,:]) over Dd columns
__global__ __launch_bounds__(256)
void dot_accum(const float* __restrict__ a, const float* __restrict__ b,
               float* __restrict__ out, int Dd, float coeff) {
    __shared__ float red[8];
    int row = blockIdx.x;
    const float* ar = a + (size_t)row * Dd;
    const float* br = b + (size_t)row * Dd;
    float s = 0.0f;
    for (int i = threadIdx.x; i < Dd; i += 256) s += ar[i] * br[i];
#pragma unroll
    for (int off = 16; off > 0; off >>= 1) s += __shfl_down(s, off, 32);
    int lane = threadIdx.x & 31, wv = threadIdx.x >> 5;
    if (lane == 0) red[wv] = s;
    __syncthreads();
    if (wv == 0) {
        s = (lane < 8) ? red[lane] : 0.0f;
#pragma unroll
        for (int off = 4; off > 0; off >>= 1) s += __shfl_down(s, off, 32);
        if (lane == 0) out[row] += coeff * s;
    }
}

// ---------------------------------------------------------------------------
extern "C" void kernel_launch(void* const* d_in, const int* in_sizes, int n_in,
                              void* d_out, int out_size, void* d_ws, size_t ws_size,
                              hipStream_t stream) {
    (void)in_sizes; (void)n_in; (void)out_size; (void)ws_size;

    const float* x  = (const float*)d_in[0];
    const float* u  = (const float*)d_in[1];
    const float* W1 = (const float*)d_in[2];   // [H,D]
    const float* b1 = (const float*)d_in[3];
    const float* W2 = (const float*)d_in[4];   // [H,H]
    const float* b2 = (const float*)d_in[5];
    const float* W3 = (const float*)d_in[6];   // [D,H]
    const float* b3 = (const float*)d_in[7];

    float* y_out  = (float*)d_out;                       // [B,D]
    float* ld_out = (float*)d_out + (size_t)BQ * DQ;     // [B]

    // Workspace layout
    char* ws = (char*)d_ws;
    size_t off = 0;
    auto alloc = [&](size_t bytes) -> char* {
        char* p = ws + off;
        off += (bytes + 255) & ~(size_t)255;
        return p;
    };
    __bf16* W1bf = (__bf16*)alloc((size_t)HQ * DQ * 2);  // [H,D]
    __bf16* W2bf = (__bf16*)alloc((size_t)HQ * HQ * 2);  // [H,H]
    __bf16* W3bf = (__bf16*)alloc((size_t)DQ * HQ * 2);  // [D,H]
    __bf16* W1T  = (__bf16*)alloc((size_t)DQ * HQ * 2);  // [D,H] = W1^T
    __bf16* W2T  = (__bf16*)alloc((size_t)HQ * HQ * 2);  // [H,H] = W2^T
    __bf16* W3T  = (__bf16*)alloc((size_t)HQ * DQ * 2);  // [H,D] = W3^T
    __bf16* xbf  = (__bf16*)alloc((size_t)BQ * DQ * 2);  // [B,D]
    __bf16* utbf = (__bf16*)alloc((size_t)BQ * DQ * 2);  // [B,D]
    __bf16* actA = (__bf16*)alloc((size_t)BQ * HQ * 2);  // h1 / s1
    __bf16* actB = (__bf16*)alloc((size_t)BQ * HQ * 2);  // h2 / s2
    float*  utF  = (float*)alloc((size_t)BQ * DQ * 4);   // [B,D]
    unsigned char* mask1 = (unsigned char*)alloc((size_t)BQ * HQ);
    unsigned char* mask2 = (unsigned char*)alloc((size_t)BQ * HQ);

    dim3 b256(256);

    // 1) fp32 -> bf16 conversions
    convert_f32_bf16<<<2048, b256, 0, stream>>>(W1, W1bf, (size_t)HQ * DQ);
    convert_f32_bf16<<<2048, b256, 0, stream>>>(W2, W2bf, (size_t)HQ * HQ);
    convert_f32_bf16<<<2048, b256, 0, stream>>>(W3, W3bf, (size_t)DQ * HQ);
    convert_f32_bf16<<<2048, b256, 0, stream>>>(x,  xbf,  (size_t)BQ * DQ);
    convert_f32_bf16<<<2048, b256, 0, stream>>>(u,  utbf, (size_t)BQ * DQ);  // ut = u

    // 2) transposed bf16 weights for the VJP direction
    dim3 tb(32, 8);
    transpose_f32_to_bf16<<<dim3(DQ / 32, HQ / 32), tb, 0, stream>>>(W1, W1T, HQ, DQ);
    transpose_f32_to_bf16<<<dim3(HQ / 32, HQ / 32), tb, 0, stream>>>(W2, W2T, HQ, HQ);
    transpose_f32_to_bf16<<<dim3(HQ / 32, DQ / 32), tb, 0, stream>>>(W3, W3T, DQ, HQ);

    // 3) forward pass
    dim3 gH(HQ / BN, BQ / BM);  // [B,H] outputs
    dim3 gD(DQ / BN, BQ / BM);  // [B,D] outputs
    // h1 = relu(x W1^T + b1), mask1
    gemm_bf16_wmma<<<gH, GEMM_THREADS, 0, stream>>>(
        xbf, W1bf, BQ, HQ, DQ, b1, nullptr, nullptr, mask1, actA, nullptr, 0);
    // h2 = relu(h1 W2^T + b2), mask2
    gemm_bf16_wmma<<<gH, GEMM_THREADS, 0, stream>>>(
        actA, W2bf, BQ, HQ, HQ, b2, nullptr, nullptr, mask2, actB, nullptr, 0);
    // y = h2 W3^T + b3 + x
    gemm_bf16_wmma<<<gD, GEMM_THREADS, 0, stream>>>(
        actB, W3bf, BQ, DQ, HQ, b3, x, nullptr, nullptr, nullptr, y_out, 1);

    // 4) logdet accumulator init
    zero_f32<<<(BQ + 255) / 256, b256, 0, stream>>>(ld_out, BQ);

    // 5) Neumann series: ut <- J^T ut, logdet += coeff[n] * <ut, u>
    for (int n = 0; n < 10; ++n) {
        // s1 = (ut @ W3) * mask2   : Bt = W3^T [H,D]
        gemm_bf16_wmma<<<gH, GEMM_THREADS, 0, stream>>>(
            utbf, W3T, BQ, HQ, DQ, nullptr, nullptr, mask2, nullptr, actA, nullptr, 2);
        // s2 = (s1 @ W2) * mask1   : Bt = W2^T [H,H]
        gemm_bf16_wmma<<<gH, GEMM_THREADS, 0, stream>>>(
            actA, W2T, BQ, HQ, HQ, nullptr, nullptr, mask1, nullptr, actB, nullptr, 2);
        // ut = s2 @ W1             : Bt = W1^T [D,H]
        gemm_bf16_wmma<<<gD, GEMM_THREADS, 0, stream>>>(
            actB, W1T, BQ, DQ, HQ, nullptr, nullptr, nullptr, nullptr, utbf, utF, 3);
        float coeff = ((n & 1) ? -1.0f : 1.0f) / (float)(n + 1);
        dot_accum<<<BQ, b256, 0, stream>>>(utF, u, ld_out, DQ, coeff);
    }
}
